// MHA_26792005992553
// MI455X (gfx1250) — matrix operand order repaired and checked by reading, not compiled, not run on hardware
//
#include <hip/hip_runtime.h>

// MHA: B=2, T=2048, DIM=1024, 16 heads x head_dim 64. fp32 in/out,
// f16 WMMA (v_wmma_f32_16x16x32_f16) with f32 accumulation throughout.

#define DIM_  1024
#define NHEAD 16
#define HDIM  64
#define BB    2
#define TT    2048
#define MROWS (BB * TT)   // 4096

typedef __attribute__((ext_vector_type(16))) _Float16 v16h;
typedef __attribute__((ext_vector_type(8)))  _Float16 v8h;
typedef __attribute__((ext_vector_type(4)))  _Float16 v4h;
typedef __attribute__((ext_vector_type(8)))  float    v8f;

union AFrag { v16h v; v8h h[2]; };

// ---------------------------------------------------------------- convert
__global__ __launch_bounds__(256)
void cvt_f32_f16(const float* __restrict__ s, _Float16* __restrict__ d, int n)
{
    int i = (blockIdx.x * 256 + threadIdx.x) * 4;
    if (i < n) {
        float4 v = *(const float4*)(s + i);
        v4h o;
        o[0] = (_Float16)v.x; o[1] = (_Float16)v.y;
        o[2] = (_Float16)v.z; o[3] = (_Float16)v.w;
        *(v4h*)(d + i) = o;
    }
}

// ---------------------------------------------------------------- GEMM
// out[m,n] = (sum_k A[m,k] * W[n,k] + bias[n]) * scale
// A: [M,K] f16 row-major, W: [N,K] f16 row-major (i.e. computes A @ W^T).
// layout 0: f16 out, head-major  [B,H,T,HD]   (for Q, K)
// layout 1: f16 out, transposed  [B,H,HD,T]   (for V)
// layout 2: f32 out, row-major   [M,N]        (final projection -> d_out)
__global__ __launch_bounds__(256)
void gemm_wmma(const _Float16* __restrict__ A, const _Float16* __restrict__ W,
               const float* __restrict__ bias, void* __restrict__ outp,
               int M, int N, int K, int layout, float scale)
{
    __shared__ __align__(16) _Float16 As[128][72];
    __shared__ __align__(16) _Float16 Bs[128][72];

    const int tid = threadIdx.x;
    const int w   = tid >> 5;
    const int l   = tid & 31;
    const int l16 = l & 15;
    const int kh  = l >> 4;          // which K-half this lane covers
    const int wm  = (w & 3) * 32;    // wave row offset in block tile
    const int wn  = (w >> 2) * 64;   // wave col offset in block tile
    const int m0  = blockIdx.y * 128;
    const int n0  = blockIdx.x * 128;

    v8f acc[2][4];
#pragma unroll
    for (int i = 0; i < 2; ++i)
#pragma unroll
        for (int j = 0; j < 4; ++j) acc[i][j] = (v8f)0.0f;

    const int lr = tid >> 1;         // 0..127: tile row loaded by this thread
    const int lc = (tid & 1) * 32;   // 32-half segment within BK=64

    for (int k0 = 0; k0 < K; k0 += 64) {
        __syncthreads();
        const _Float16* ga = A + (size_t)(m0 + lr) * K + k0 + lc;
        const _Float16* gb = W + (size_t)(n0 + lr) * K + k0 + lc;
#pragma unroll
        for (int j = 0; j < 4; ++j) {
            *(v8h*)&As[lr][lc + 8 * j] = *(const v8h*)(ga + 8 * j);
            *(v8h*)&Bs[lr][lc + 8 * j] = *(const v8h*)(gb + 8 * j);
        }
        if (k0 + 64 < K) {           // hint next K-slab into cache
            __builtin_prefetch(ga + 64, 0, 1);
            __builtin_prefetch(gb + 64, 0, 1);
        }
        __syncthreads();

#pragma unroll
        for (int ks = 0; ks < 64; ks += 32) {
            AFrag a[2];
#pragma unroll
            for (int ti = 0; ti < 2; ++ti) {
                const _Float16* pa = &As[wm + ti * 16 + l16][ks + 16 * kh];
                a[ti].h[0] = *(const v8h*)pa;
                a[ti].h[1] = *(const v8h*)(pa + 8);
            }
#pragma unroll
            for (int tj = 0; tj < 4; ++tj) {
                v16h b = *(const v16h*)&Bs[wn + tj * 16 + l16][ks + 16 * kh];
#pragma unroll
                for (int ti = 0; ti < 2; ++ti)
                    acc[ti][tj] = __builtin_amdgcn_wmma_f32_16x16x32_f16(
                        false, a[ti].v, false, b, (short)0, acc[ti][tj],
                        false, false);
            }
        }
    }

    // epilogue — C/D layout: VGPR r, lane l -> row r + 8*(l>>4), col l&15
#pragma unroll
    for (int ti = 0; ti < 2; ++ti) {
#pragma unroll
        for (int tj = 0; tj < 4; ++tj) {
            const int n  = n0 + wn + tj * 16 + l16;
            const float bv = bias[n];
#pragma unroll
            for (int r = 0; r < 8; ++r) {
                const int m = m0 + wm + ti * 16 + 8 * kh + r;
                const float val = (acc[ti][tj][r] + bv) * scale;
                if (layout == 2) {
                    ((float*)outp)[(size_t)m * N + n] = val;
                } else {
                    const int b = m >> 11, t = m & (TT - 1);
                    const int h = n >> 6,  d = n & (HDIM - 1);
                    size_t idx;
                    if (layout == 0)
                        idx = (((size_t)(b * NHEAD + h) * TT + t) * HDIM) + d;
                    else
                        idx = (((size_t)(b * NHEAD + h) * HDIM + d) * TT) + t;
                    ((_Float16*)outp)[idx] = (_Float16)val;
                }
            }
        }
    }
}

// ---------------------------------------------------------------- attention
// Flash-attention. Grid: (T/128, B*NHEAD). 8 waves, each owns 16 query rows.
// Qh, Kh: [B,H,T,64] f16 (Q pre-scaled by 1/8). Vt: [B,H,64,T] f16.
// Ah out: [B*T, DIM] f16 row-major (ready to feed the output projection).
__global__ __launch_bounds__(256)
void attn_wmma(const _Float16* __restrict__ Qh, const _Float16* __restrict__ Kh,
               const _Float16* __restrict__ Vt, _Float16* __restrict__ Ah)
{
    __shared__ __align__(16) _Float16 Qs[128][72];
    __shared__ __align__(16) _Float16 Ks[64][72];
    __shared__ __align__(16) _Float16 Vts[64][72];      // [d][kv]
    __shared__ __align__(16) _Float16 Ps[8][16][72];    // per-wave P tile

    const int tid = threadIdx.x;
    const int w   = tid >> 5;
    const int l   = tid & 31;
    const int l16 = l & 15;
    const int kh  = l >> 4;
    const int q0  = blockIdx.x * 128;
    const int bh  = blockIdx.y;                  // b*NHEAD + h
    const size_t headQK = (size_t)bh * TT * HDIM;
    const size_t headV  = (size_t)bh * HDIM * TT;

    {   // stage Q tile (128 x 64)
        const int row = tid >> 1, seg = (tid & 1) * 32;
        const _Float16* gq = Qh + headQK + (size_t)(q0 + row) * HDIM + seg;
#pragma unroll
        for (int j = 0; j < 4; ++j)
            *(v8h*)&Qs[row][seg + 8 * j] = *(const v8h*)(gq + 8 * j);
    }

    float m_i[8], l_i[8];
    v8f o[4];
#pragma unroll
    for (int r = 0; r < 8; ++r) { m_i[r] = -1e30f; l_i[r] = 0.0f; }
#pragma unroll
    for (int j = 0; j < 4; ++j) o[j] = (v8f)0.0f;

    const int wq = w * 16;                       // wave's query rows in Qs

    for (int kv0 = 0; kv0 < TT; kv0 += 64) {
        __syncthreads();
        {   // stage K (64 x 64) and V^T (64 x 64)
            const int row = tid >> 2, seg = (tid & 3) * 16;
            const _Float16* gk = Kh + headQK + (size_t)(kv0 + row) * HDIM + seg;
            *(v8h*)&Ks[row][seg]     = *(const v8h*)gk;
            *(v8h*)&Ks[row][seg + 8] = *(const v8h*)(gk + 8);
            const _Float16* gv = Vt + headV + (size_t)row * TT + kv0 + seg;
            *(v8h*)&Vts[row][seg]     = *(const v8h*)gv;
            *(v8h*)&Vts[row][seg + 8] = *(const v8h*)(gv + 8);
        }
        __syncthreads();

        // S = Q . K^T   (K-dim = head dim, two 32-wide WMMA steps)
        v8f s[4];
#pragma unroll
        for (int j = 0; j < 4; ++j) s[j] = (v8f)0.0f;
#pragma unroll
        for (int ks = 0; ks < 64; ks += 32) {
            AFrag a;
            const _Float16* pa = &Qs[wq + l16][ks + 16 * kh];
            a.h[0] = *(const v8h*)pa;
            a.h[1] = *(const v8h*)(pa + 8);
#pragma unroll
            for (int tj = 0; tj < 4; ++tj) {
                v16h b = *(const v16h*)&Ks[tj * 16 + l16][ks + 16 * kh];
                s[tj] = __builtin_amdgcn_wmma_f32_16x16x32_f16(
                    false, a.v, false, b, (short)0, s[tj], false, false);
            }
        }

        // online softmax over this kv tile; row = r + 8*kh, 16 cols per lane grp
        float p[4][8];
#pragma unroll
        for (int r = 0; r < 8; ++r) {
            float rmax = fmaxf(fmaxf(s[0][r], s[1][r]), fmaxf(s[2][r], s[3][r]));
#pragma unroll
            for (int off = 1; off < 16; off <<= 1)
                rmax = fmaxf(rmax, __shfl_xor(rmax, off, 32));
            const float mn    = fmaxf(m_i[r], rmax);
            const float alpha = __expf(m_i[r] - mn);
            float rs = 0.0f;
#pragma unroll
            for (int tj = 0; tj < 4; ++tj) {
                p[tj][r] = __expf(s[tj][r] - mn);
                rs += p[tj][r];
            }
#pragma unroll
            for (int off = 1; off < 16; off <<= 1)
                rs += __shfl_xor(rs, off, 32);
            l_i[r] = l_i[r] * alpha + rs;
            m_i[r] = mn;
#pragma unroll
            for (int tj = 0; tj < 4; ++tj) o[tj][r] *= alpha;
        }

        // spill P (C-layout) to wave-private LDS, reload as A-fragments
#pragma unroll
        for (int tj = 0; tj < 4; ++tj)
#pragma unroll
            for (int r = 0; r < 8; ++r)
                Ps[w][r + 8 * kh][tj * 16 + l16] = (_Float16)p[tj][r];
        __syncthreads();

        // O += P . V   (K-dim = kv, two 32-wide WMMA steps)
#pragma unroll
        for (int kk = 0; kk < 64; kk += 32) {
            AFrag a;
            const _Float16* pa = &Ps[w][l16][kk + 16 * kh];
            a.h[0] = *(const v8h*)pa;
            a.h[1] = *(const v8h*)(pa + 8);
#pragma unroll
            for (int tj = 0; tj < 4; ++tj) {
                v16h b = *(const v16h*)&Vts[tj * 16 + l16][kk + 16 * kh];
                o[tj] = __builtin_amdgcn_wmma_f32_16x16x32_f16(
                    false, a.v, false, b, (short)0, o[tj], false, false);
            }
        }
    }

    // normalize and store into [B*T, DIM] f16 for the output projection
    const int b = bh >> 4, h = bh & (NHEAD - 1);
#pragma unroll
    for (int r = 0; r < 8; ++r) {
        const float inv = 1.0f / l_i[r];
        const int t = q0 + wq + r + 8 * kh;
#pragma unroll
        for (int tj = 0; tj < 4; ++tj) {
            const float val = o[tj][r] * inv;
            Ah[(size_t)(b * TT + t) * DIM_ + h * HDIM + tj * 16 + l16] =
                (_Float16)val;
        }
    }
}

// ---------------------------------------------------------------- launch
extern "C" void kernel_launch(void* const* d_in, const int* in_sizes, int n_in,
                              void* d_out, int out_size, void* d_ws, size_t ws_size,
                              hipStream_t stream)
{
    (void)in_sizes; (void)n_in; (void)out_size; (void)ws_size;

    const float* x  = (const float*)d_in[0];
    const float* Wq = (const float*)d_in[1];
    const float* bq = (const float*)d_in[2];
    const float* Wk = (const float*)d_in[3];
    const float* bk = (const float*)d_in[4];
    const float* Wv = (const float*)d_in[5];
    const float* bv = (const float*)d_in[6];
    const float* Wo = (const float*)d_in[7];
    const float* bo = (const float*)d_in[8];

    char* ws = (char*)d_ws;
    const size_t szX = (size_t)MROWS * DIM_ * sizeof(_Float16);   // 8 MiB
    const size_t szW = (size_t)DIM_  * DIM_ * sizeof(_Float16);   // 2 MiB

    _Float16* xh  = (_Float16*)ws;            ws += szX;
    _Float16* wqh = (_Float16*)ws;            ws += szW;
    _Float16* wkh = (_Float16*)ws;            ws += szW;
    _Float16* wvh = (_Float16*)ws;            ws += szW;
    _Float16* woh = (_Float16*)ws;            ws += szW;
    _Float16* Qh  = (_Float16*)ws;            ws += szX;
    _Float16* KhB = (_Float16*)ws;            ws += szX;
    _Float16* VtB = (_Float16*)ws;            ws += szX;
    _Float16* AhB = (_Float16*)ws;            ws += szX;   // total 48 MiB

    const int nX = MROWS * DIM_;
    const int nW = DIM_ * DIM_;
    cvt_f32_f16<<<dim3((nX / 4 + 255) / 256), 256, 0, stream>>>(x,  xh,  nX);
    cvt_f32_f16<<<dim3((nW / 4 + 255) / 256), 256, 0, stream>>>(Wq, wqh, nW);
    cvt_f32_f16<<<dim3((nW / 4 + 255) / 256), 256, 0, stream>>>(Wk, wkh, nW);
    cvt_f32_f16<<<dim3((nW / 4 + 255) / 256), 256, 0, stream>>>(Wv, wvh, nW);
    cvt_f32_f16<<<dim3((nW / 4 + 255) / 256), 256, 0, stream>>>(Wo, woh, nW);

    const dim3 gGemm(DIM_ / 128, MROWS / 128);   // 8 x 32
    const float qscale = 0.125f;                 // 1/sqrt(HDIM)
    gemm_wmma<<<gGemm, 256, 0, stream>>>(xh, wqh, bq, Qh,  MROWS, DIM_, DIM_, 0, qscale);
    gemm_wmma<<<gGemm, 256, 0, stream>>>(xh, wkh, bk, KhB, MROWS, DIM_, DIM_, 0, 1.0f);
    gemm_wmma<<<gGemm, 256, 0, stream>>>(xh, wvh, bv, VtB, MROWS, DIM_, DIM_, 1, 1.0f);

    attn_wmma<<<dim3(TT / 128, BB * NHEAD), 256, 0, stream>>>(Qh, KhB, VtB, AhB);

    gemm_wmma<<<gGemm, 256, 0, stream>>>(AhB, woh, bo, d_out, MROWS, DIM_, DIM_, 2, 1.0f);
}
